// PredCorrNetwork_57389353009416
// MI455X (gfx1250) — matrix operand order repaired and checked by reading, not compiled, not run on hardware
//
#include <hip/hip_runtime.h>
#include <hip/hip_bf16.h>

// ---------------- problem constants ----------------
#define DIMQ      64
#define NPAR      4096
#define NSTEPS    20
#define NSUB      5
#define FORCE_MAG 5.0f
#define GAMMA_C   0.99f
#define C1_C      0.8f
#define C2_C      0.01f
#define C3_C      5.0f      // 1.0 * N_SUB

// ---------------- output float offsets (concat of reference tuple) ----------
#define OUT_X    0
#define OUT_Y    (NPAR * DIMQ)                 // 262144
#define OUT_LOSS (2 * NPAR * DIMQ)            // 524288
#define OUT_FID  (OUT_LOSS + 1)               // 524289
#define OUT_NS   (OUT_FID + NSTEPS * NPAR)    // 606209
#define OUT_ACT  (OUT_NS + NSTEPS * NPAR)     // 688129

// ---------------- workspace byte offsets ----------------
#define WS_SW0   0u          // 128x512 bf16 packed  = 131072 B
#define WS_SW1   131072u     // 512x256              = 262144 B
#define WS_SW2   393216u     // 256x256              = 131072 B
#define WS_SW3   524288u     // 256x64               =  32768 B
#define WS_AW1   557056u     // 128x64               =  16384 B
#define WS_CW0   573440u     // 64x64                =   8192 B
#define WS_CW1   581632u     // 64x32                =   4096 B
#define WS_CW2   585728u     // 32x1 -> pad 16       =   1024 B
#define WS_H0    586752u     // f32 packed H0dt^T    =  16384 B
#define WS_H1    603136u     // f32 packed H1dt^T    =  16384 B
#define WS_PART  619520u     // 256 f32 partial losses

// ---------------- LDS layout (one 16-trajectory tile per 4-wave workgroup) --
#define SHARED_BYTES 33536   // 32768 (packed H0,H1) + 768 (targets/n_op)
// arena offsets (bytes), single arena per workgroup
#define A_XS   0
#define A_YS   4096
#define A_XT   8192
#define A_YT   12288
#define A_IN   16384   // 16x128 bf16
#define A_B1   20480   // 16x512 bf16
#define A_B2   36864   // 16x256 bf16
#define A_B3   45056   // 16x256 bf16
#define A_BS   53248   // 16x64  bf16
#define A_BA   55296   // 16x128 bf16
#define A_BC1  59392   // 16x64  bf16
#define A_BC2  61440   // 16x32  bf16
#define A_AL   62464   // 16 f32 alpha
#define ARENA_BYTES 62528
#define SMEM_TOTAL (SHARED_BYTES + ARENA_BYTES)   // 96064

typedef float  v2f   __attribute__((ext_vector_type(2)));
typedef float  v8f   __attribute__((ext_vector_type(8)));
typedef __bf16 v16bf __attribute__((ext_vector_type(16)));
typedef unsigned int uv4 __attribute__((ext_vector_type(4)));

union BFrag { uv4 q[2]; v16bf v; };

// ---------------- WMMA wrappers ----------------
__device__ __forceinline__ v8f wmma_bf16(const BFrag& a, const BFrag& b, v8f c) {
    return __builtin_amdgcn_wmma_f32_16x16x32_bf16(false, a.v, false, b.v,
                                                   (short)0, c, false, false);
}

__device__ __forceinline__ v8f wmma_f32k4(v2f a, v2f b, v8f c) {
#if __has_builtin(__builtin_amdgcn_wmma_f32_16x16x4_f32)
    return __builtin_amdgcn_wmma_f32_16x16x4_f32(false, a, false, b,
                                                 (short)0, c, false, false);
#else
    c[0] += a[0] * b[0];
    return c;
#endif
}

// ---------------- bf16 GEMM layer (cooperative: wave `wid` does every 4th
// N-tile): D(16xNout) = relu(A(16xK) @ Wp + bias)  [optionally += prior dst]
// src: LDS row-major bf16 (row stride K, K multiple of 32)
// Wp : global packed B fragments [KT][NT][lane 32][16 bf16]
__device__ __forceinline__ void gemm_bf16_layer(
    const __bf16* __restrict__ src, int K,
    const __bf16* __restrict__ Wp, int KT, int NT, int Nout,
    const float* __restrict__ bias, __bf16* __restrict__ dst,
    int lane, int wid, bool addto)
{
    const int h = lane >> 4, nl = lane & 15;
    for (int nt = wid; nt < NT; nt += 4) {   // wid is SGPR -> scalar loop
        v8f acc = {0.f,0.f,0.f,0.f,0.f,0.f,0.f,0.f};
        for (int kt = 0; kt < KT; ++kt) {
            BFrag a, b;
            const __bf16* ap = src + nl * K + kt * 32 + h * 8;
            a.q[0] = *(const uv4*)(ap);
            a.q[1] = *(const uv4*)(ap + 16);
            const __bf16* bp = Wp + (((size_t)(kt * NT + nt) * 32 + lane) << 4);
            b.q[0] = *(const uv4*)(bp);
            b.q[1] = *(const uv4*)(bp + 8);
            acc = wmma_bf16(a, b, acc);
        }
        const float bv = bias[nt * 16 + nl];
        const int ncol = nt * 16 + nl;
#pragma unroll
        for (int r = 0; r < 8; ++r) {
            float v = fmaxf(acc[r] + bv, 0.0f);
            const int idx = (r + 8 * h) * Nout + ncol;
            if (addto) v += (float)dst[idx];
            dst[idx] = (__bf16)v;
        }
    }
}

// ---------------- f32 physics GEMM, single n-tile:
// F = sgn*(A@H0T + alpha .* A@H1T) for columns [nt*16, nt*16+16)
// src: LDS 16x64 f32 row-major; H0l/H1l: LDS packed [kt 16][nt 4][lane 32][2]
__device__ __forceinline__ v8f phys_gemm1(
    const float* __restrict__ src,
    const float* __restrict__ H0l, const float* __restrict__ H1l,
    int lane, int nt, v8f avv, float sgn)
{
    const int h = lane >> 4, nl = lane & 15;
    v8f a0 = {0.f,0.f,0.f,0.f,0.f,0.f,0.f,0.f};
    v8f a1 = {0.f,0.f,0.f,0.f,0.f,0.f,0.f,0.f};
#pragma unroll
    for (int kt = 0; kt < 16; ++kt) {
        v2f a  = *(const v2f*)(src + nl * 64 + kt * 4 + h * 2);
        const int bo = ((kt * 4 + nt) * 32 + lane) * 2;
        v2f b0 = *(const v2f*)(H0l + bo);
        v2f b1 = *(const v2f*)(H1l + bo);
        a0 = wmma_f32k4(a, b0, a0);
        a1 = wmma_f32k4(a, b1, a1);
    }
    v8f F;
#pragma unroll
    for (int r = 0; r < 8; ++r)
        F[r] = sgn * (a0[r] + avv[r] * a1[r]);
    return F;
}

// ---------------- prep: pack f32 weights -> bf16 WMMA B-fragment order ------
__global__ void pack_w(const float* __restrict__ W, int K, int N, int Npad,
                       __bf16* __restrict__ out)
{
    const int NT = Npad >> 4;
    const int total = (K >> 5) * NT * 256;
    int id = blockIdx.x * blockDim.x + threadIdx.x;
    if (id >= total) return;
    int v = id & 7; int t = id >> 3;
    int lane = t & 31; t >>= 5;
    int nt = t % NT, kt = t / NT;
    int h = lane >> 4, nl = lane & 15;
    int n = nt * 16 + nl;
    int k = kt * 32 + h * 8 + v * 2 + (v >> 2) * 8;
    float f0 = (n < N) ? W[(size_t)k * N + n]       : 0.f;
    float f1 = (n < N) ? W[(size_t)(k + 1) * N + n] : 0.f;
    out[(size_t)id * 2]     = (__bf16)f0;
    out[(size_t)id * 2 + 1] = (__bf16)f1;
}

// ---------------- prep: pack H^T (f32) into f32 16x16x4 B-fragment order ----
__global__ void pack_h(const float* __restrict__ H, float* __restrict__ out)
{
    int id = blockIdx.x * blockDim.x + threadIdx.x;   // [kt16][nt4][lane32][v2]
    if (id >= 4096) return;
    int v = id & 1, lane = (id >> 1) & 31, nt = (id >> 6) & 3, kt = id >> 8;
    int h = lane >> 4, nl = lane & 15;
    int n = nt * 16 + nl;
    int k = kt * 4 + h * 2 + v;
    out[id] = H[n * 64 + k];   // (H^T)[k][n]
}

// ---------------- deterministic loss reduction ----------------
__global__ void reduce_loss(const float* __restrict__ partial, float* __restrict__ out)
{
    if (blockIdx.x == 0 && threadIdx.x == 0) {
        double s = 0.0;
        for (int i = 0; i < 256; ++i) s += (double)partial[i];
        out[OUT_LOSS] = (float)(s / (double)NPAR);
    }
}

// ---------------- fused main kernel: 4 waves cooperate on one 16-traj tile --
__global__ __launch_bounds__(128) void pcn_main(
    const float* __restrict__ psi_x, const float* __restrict__ psi_y,
    const float* __restrict__ tgx_g, const float* __restrict__ tgy_g,
    const float* __restrict__ nop_g,
    const float* __restrict__ sb0, const float* __restrict__ sb1,
    const float* __restrict__ sb2, const float* __restrict__ sb3,
    const float* __restrict__ aW0, const float* __restrict__ ab0,
    const float* __restrict__ ab1,
    const float* __restrict__ cb0, const float* __restrict__ cb1,
    const float* __restrict__ cb2,
    const char* __restrict__ ws, float* __restrict__ partial,
    float* __restrict__ out)
{
    extern __shared__ char smem[];
    const int tid  = threadIdx.x;
    const int lane = tid & 31;
    // wave id is uniform across the wave: pin it to an SGPR so the N-tile
    // loops become scalar branches and EXEC stays statically all-1s (WMMA req)
    const int wid  = __builtin_amdgcn_readfirstlane(tid >> 5);
    const int tile = blockIdx.x;                 // one 16-traj tile per WG
    const int base = tile * 16;
    const int h = lane >> 4, nl = lane & 15;

    float* Hl = (float*)smem;                 // 8192 f32: [H0 4096][H1 4096]
    float* tg = (float*)(smem + 32768);       // 192 f32: [tx 64][ty 64][nop 64]

    char* wb = smem + SHARED_BYTES;
    float*  Xs     = (float*)(wb + A_XS);
    float*  Ys     = (float*)(wb + A_YS);
    float*  XT     = (float*)(wb + A_XT);
    float*  YT     = (float*)(wb + A_YT);
    __bf16* bufIn  = (__bf16*)(wb + A_IN);
    __bf16* buf1   = (__bf16*)(wb + A_B1);
    __bf16* buf2   = (__bf16*)(wb + A_B2);
    __bf16* buf3   = (__bf16*)(wb + A_B3);
    __bf16* bufS   = (__bf16*)(wb + A_BS);
    __bf16* bufA   = (__bf16*)(wb + A_BA);
    __bf16* bufC1  = (__bf16*)(wb + A_BC1);
    __bf16* bufC2  = (__bf16*)(wb + A_BC2);
    float*  alphaA = (float*)(wb + A_AL);

    const __bf16* sW0p = (const __bf16*)(ws + WS_SW0);
    const __bf16* sW1p = (const __bf16*)(ws + WS_SW1);
    const __bf16* sW2p = (const __bf16*)(ws + WS_SW2);
    const __bf16* sW3p = (const __bf16*)(ws + WS_SW3);
    const __bf16* aW1p = (const __bf16*)(ws + WS_AW1);
    const __bf16* cW0p = (const __bf16*)(ws + WS_CW0);
    const __bf16* cW1p = (const __bf16*)(ws + WS_CW1);
    const __bf16* cW2p = (const __bf16*)(ws + WS_CW2);
    const float*  H0p  = (const float*)(ws + WS_H0);
    const float*  H1p  = (const float*)(ws + WS_H1);

    // stage shared packed Hamiltonians + targets + initial state
    for (int i = tid; i < 8192; i += 128)
        Hl[i] = (i < 4096) ? H0p[i] : H1p[i - 4096];
    for (int i = tid; i < 192; i += 128)
        tg[i] = (i < 64) ? tgx_g[i] : (i < 128) ? tgy_g[i - 64] : nop_g[i - 128];
    for (int e = tid; e < 1024; e += 128) {
        int m = e >> 6, k = e & 63;
        Xs[m * 64 + k] = psi_x[(base + m) * 64 + k];
        Ys[m * 64 + k] = psi_y[(base + m) * 64 + k];
    }
    if (tid < 16) alphaA[tid] = 0.0f;
    __syncthreads();

    float g = 1.0f, lossm = 0.0f, lastFid = 0.0f;

    for (int t = 0; t < NSTEPS; ++t) {
        // ------- build NN input [x;y] bf16 (16x128) + action layer 0 -------
        for (int e = tid; e < 1024; e += 128) {
            int m = e >> 6, kp = (e & 63) << 1;
            float v0, v1;
            if (kp < 64) { v0 = Xs[m * 64 + kp];      v1 = Xs[m * 64 + kp + 1]; }
            else         { v0 = Ys[m * 64 + kp - 64]; v1 = Ys[m * 64 + kp - 63]; }
            bufIn[m * 128 + kp]     = (__bf16)v0;
            bufIn[m * 128 + kp + 1] = (__bf16)v1;
            // action net layer 0 uses previous-step alpha (scalar -> 128)
            int j = kp;
            float al = alphaA[m];
            bufA[m * 128 + j]     = (__bf16)fmaxf(al * aW0[j]     + ab0[j],     0.f);
            bufA[m * 128 + j + 1] = (__bf16)fmaxf(al * aW0[j + 1] + ab0[j + 1], 0.f);
        }
        __syncthreads();
        // ------- state net (N split across 4 waves, barrier per layer) ------
        gemm_bf16_layer(bufIn, 128, sW0p,  4, 32, 512, sb0, buf1, lane, wid, false);
        __syncthreads();
        gemm_bf16_layer(buf1,  512, sW1p, 16, 16, 256, sb1, buf2, lane, wid, false);
        __syncthreads();
        gemm_bf16_layer(buf2,  256, sW2p,  8, 16, 256, sb2, buf3, lane, wid, false);
        __syncthreads();
        gemm_bf16_layer(buf3,  256, sW3p,  8,  4,  64, sb3, bufS, lane, wid, false);
        // action layer 1 accumulates into the same (wave-owned) bufS elements
        gemm_bf16_layer(bufA, 128, aW1p, 4, 4, 64, ab1, bufS, lane, wid, true);
        __syncthreads();
        // ------- combine net -------
        gemm_bf16_layer(bufS,  64, cW0p, 2, 4, 64, cb0, bufC1, lane, wid, false);
        __syncthreads();
        gemm_bf16_layer(bufC1, 64, cW1p, 2, 2, 32, cb1, bufC2, lane, wid, false);
        __syncthreads();
        if (wid == 0) { // final 32 -> 1 (padded to N=16, only column 0 valid)
            BFrag a, b;
            const __bf16* ap = bufC2 + nl * 32 + h * 8;
            a.q[0] = *(const uv4*)(ap);
            a.q[1] = *(const uv4*)(ap + 16);
            const __bf16* bp = cW2p + lane * 16;
            b.q[0] = *(const uv4*)(bp);
            b.q[1] = *(const uv4*)(bp + 8);
            v8f acc = {0.f,0.f,0.f,0.f,0.f,0.f,0.f,0.f};
            acc = wmma_bf16(a, b, acc);
            if (nl == 0) {
                float cb = cb2[0];
#pragma unroll
                for (int r = 0; r < 8; ++r) {
                    float av_ = acc[r] + cb;
                    av_ = fminf(fmaxf(av_, -FORCE_MAG), FORCE_MAG);
                    alphaA[r + 8 * h] = av_;
                }
            }
        }
        __syncthreads();
        // per-lane alpha broadcast for the C-layout rows this lane owns
        v8f avv;
#pragma unroll
        for (int r = 0; r < 8; ++r) avv[r] = alphaA[r + 8 * h];

        // ------- physics: NSUB Heun sub-steps, n-tile `wid` per wave -------
        for (int s = 0; s < NSUB; ++s) {
            v8f Fx = phys_gemm1(Ys, Hl, Hl + 4096, lane, wid, avv,  1.0f); //  H@y
            v8f Fy = phys_gemm1(Xs, Hl, Hl + 4096, lane, wid, avv, -1.0f); // -H@x
#pragma unroll
            for (int r = 0; r < 8; ++r) {
                int idx = (r + 8 * h) * 64 + wid * 16 + nl;
                XT[idx] = Xs[idx] + Fx[r];
                YT[idx] = Ys[idx] + Fy[r];
            }
            __syncthreads();
            v8f Gx = phys_gemm1(YT, Hl, Hl + 4096, lane, wid, avv,  1.0f); //  H@yt
#pragma unroll
            for (int r = 0; r < 8; ++r) {
                int idx = (r + 8 * h) * 64 + wid * 16 + nl;
                Xs[idx] = Xs[idx] + 0.5f * (Gx[r] + Fx[r]);
            }
            v8f Gy = phys_gemm1(XT, Hl, Hl + 4096, lane, wid, avv, -1.0f); // -(H@xt)
#pragma unroll
            for (int r = 0; r < 8; ++r) {
                int idx = (r + 8 * h) * 64 + wid * 16 + nl;
                Ys[idx] = Ys[idx] + 0.5f * (Gy[r] + Fy[r]);
            }
            __syncthreads();
        }

        // ------- observables / loss (wave 0, lanes 0..15 own trajectories) --
        if (wid == 0 && lane < 16) {
            float txx = 0.f, tyy = 0.f, tyx = 0.f, txy = 0.f, mn = 0.f;
            for (int k = 0; k < 64; ++k) {
                float xv = Xs[lane * 64 + k], yv = Ys[lane * 64 + k];
                float tx = tg[k], ty = tg[64 + k];
                txx += tx * xv; tyy += ty * yv;
                tyx += ty * xv; txy += tx * yv;
                mn  += tg[128 + k] * (xv * xv + yv * yv);
            }
            float fid = txx*txx + tyy*tyy + tyx*tyx + txy*txy
                      + 2.f*txx*tyy - 2.f*txy*tyx;
            float a1 = alphaA[lane];
            lossm += C1_C * g * (1.f - fid) + g * C2_C * fabsf(a1);
            lastFid = fid;
            out[OUT_FID + t * NPAR + base + lane] = fid;
            out[OUT_NS  + t * NPAR + base + lane] = mn;
            out[OUT_ACT + t * NPAR + base + lane] = a1;
        }
        g *= GAMMA_C;
        __syncthreads();   // loss reads & next-step input reads of Xs/Ys
    }

    if (wid == 0) {
        if (lane < 16) lossm += C3_C * (1.f - lastFid);
        float l = lossm;
        for (int off = 16; off > 0; off >>= 1) l += __shfl_xor(l, off, 32);
        if (lane == 0) partial[tile] = l;
    }

    // final states
    for (int e = tid; e < 1024; e += 128) {
        int m = e >> 6, k = e & 63;
        out[OUT_X + (base + m) * 64 + k] = Xs[m * 64 + k];
        out[OUT_Y + (base + m) * 64 + k] = Ys[m * 64 + k];
    }
}

// ---------------- host launcher ----------------
extern "C" void kernel_launch(void* const* d_in, const int* in_sizes, int n_in,
                              void* d_out, int out_size, void* d_ws, size_t ws_size,
                              hipStream_t stream) {
    (void)in_sizes; (void)n_in; (void)out_size; (void)ws_size;
    const float* psi_x = (const float*)d_in[0];
    const float* psi_y = (const float*)d_in[1];
    const float* H0dt  = (const float*)d_in[2];
    const float* H1dt  = (const float*)d_in[3];
    const float* tgx   = (const float*)d_in[4];
    const float* tgy   = (const float*)d_in[5];
    const float* nop   = (const float*)d_in[6];
    const float* sW0 = (const float*)d_in[7];  const float* sb0 = (const float*)d_in[8];
    const float* sW1 = (const float*)d_in[9];  const float* sb1 = (const float*)d_in[10];
    const float* sW2 = (const float*)d_in[11]; const float* sb2 = (const float*)d_in[12];
    const float* sW3 = (const float*)d_in[13]; const float* sb3 = (const float*)d_in[14];
    const float* aW0 = (const float*)d_in[15]; const float* ab0 = (const float*)d_in[16];
    const float* aW1 = (const float*)d_in[17]; const float* ab1 = (const float*)d_in[18];
    const float* cW0 = (const float*)d_in[19]; const float* cb0 = (const float*)d_in[20];
    const float* cW1 = (const float*)d_in[21]; const float* cb1 = (const float*)d_in[22];
    const float* cW2 = (const float*)d_in[23]; const float* cb2 = (const float*)d_in[24];

    char* ws = (char*)d_ws;
    float* out = (float*)d_out;
    float* partial = (float*)(ws + WS_PART);

    auto pk = [&](const float* W, int K, int N, int Npad, unsigned off) {
        int total = (K >> 5) * (Npad >> 4) * 256;
        pack_w<<<(total + 255) / 256, 256, 0, stream>>>(W, K, N, Npad,
                                                        (__bf16*)(ws + off));
    };
    pk(sW0, 128, 512, 512, WS_SW0);
    pk(sW1, 512, 256, 256, WS_SW1);
    pk(sW2, 256, 256, 256, WS_SW2);
    pk(sW3, 256,  64,  64, WS_SW3);
    pk(aW1, 128,  64,  64, WS_AW1);
    pk(cW0,  64,  64,  64, WS_CW0);
    pk(cW1,  64,  32,  32, WS_CW1);
    pk(cW2,  32,   1,  16, WS_CW2);
    pack_h<<<16, 256, 0, stream>>>(H0dt, (float*)(ws + WS_H0));
    pack_h<<<16, 256, 0, stream>>>(H1dt, (float*)(ws + WS_H1));

    pcn_main<<<256, 128, SMEM_TOTAL, stream>>>(
        psi_x, psi_y, tgx, tgy, nop,
        sb0, sb1, sb2, sb3,
        aW0, ab0, ab1, cb0, cb1, cb2,
        (const char*)ws, partial, out);

    reduce_loss<<<1, 32, 0, stream>>>(partial, out);
}
